// MoE_2284922602128
// MI455X (gfx1250) — matrix operand order repaired and checked by reading, not compiled, not run on hardware
//
#include <hip/hip_runtime.h>
#include <hip/hip_bf16.h>
#include <math.h>

// ---------------- problem constants ----------------
#define BTOK 4096
#define NEXP 8
#define DIN  768
#define DH   2048
#define DOUT 768
#define TOPK 2
#define ROWS (BTOK*TOPK)          // 8192 selected (token,expert) rows
#define BM   64                   // block M-tile (4 WMMA m-tiles)
#define CAP  (ROWS + NEXP*BM)     // padded capacity (each expert padded to x64) = 8704
#define MTILES (CAP/BM)           // 136
#define EPS_F 2.2204460492503131e-16f

typedef __attribute__((ext_vector_type(16))) __bf16 bf16x16;
typedef __attribute__((ext_vector_type(8)))  __bf16 bf16x8;
typedef __attribute__((ext_vector_type(8)))  float  f32x8;

// ---------------- helpers ----------------
__device__ __forceinline__ float gelu_tanh(float x) {
    float x3 = x * x * x;
    return 0.5f * x * (1.0f + tanhf(0.7978845608028654f * (x + 0.044715f * x3)));
}

__device__ __forceinline__ bf16x16 load_afrag(const __bf16* p) {
    // lane<16 holds K 0..15 ; lane>=16 holds K 16..31 (contiguous, matches B)
    bf16x8 lo = *(const bf16x8*)p;
    bf16x8 hi = *(const bf16x8*)(p + 8);
    bf16x16 a;
#pragma unroll
    for (int i = 0; i < 8; ++i) { a[i] = lo[i]; a[i + 8] = hi[i]; }
    return a;
}

#define WMMA_BF16(A, B, C) \
    __builtin_amdgcn_wmma_f32_16x16x32_bf16(false, (A), false, (B), (short)0, (C), false, false)

// ---------------- 0. init ----------------
__global__ void moe_init(float* __restrict__ out, int out_n,
                         int* __restrict__ token_of_row, float* __restrict__ gate_of_row,
                         int* __restrict__ counts, int* __restrict__ cursor,
                         float* __restrict__ importance, int* __restrict__ loadcnt) {
    int stride = gridDim.x * blockDim.x;
    int i = blockIdx.x * blockDim.x + threadIdx.x;
    for (int idx = i; idx < out_n; idx += stride) out[idx] = 0.0f;
    for (int idx = i; idx < CAP; idx += stride) { token_of_row[idx] = -1; gate_of_row[idx] = 0.0f; }
    if (i < NEXP) { counts[i] = 0; cursor[i] = 0; importance[i] = 0.0f; loadcnt[i] = 0; }
}

// ---------------- 1. fp32 -> bf16 straight cast ----------------
__global__ void cast_bf16(const float* __restrict__ in, __bf16* __restrict__ out, long n) {
    long i = (long)(blockIdx.x * blockDim.x + threadIdx.x) * 4;
    long stride = (long)gridDim.x * blockDim.x * 4;
    for (; i + 3 < n; i += stride) {
        out[i+0] = (__bf16)in[i+0];
        out[i+1] = (__bf16)in[i+1];
        out[i+2] = (__bf16)in[i+2];
        out[i+3] = (__bf16)in[i+3];
    }
}

// ---------------- 2. fp32 [E][R][C] -> bf16 [E][C][R] transpose cast ----------------
__global__ void cast_bf16_t(const float* __restrict__ in, __bf16* __restrict__ out,
                            int R, int C, long n) {
    long stride = (long)gridDim.x * blockDim.x;
    for (long idx = blockIdx.x * blockDim.x + threadIdx.x; idx < n; idx += stride) {
        long rc = (long)R * C;
        int e = (int)(idx / rc);
        long rem = idx - (long)e * rc;
        int r = (int)(rem / C);
        int c = (int)(rem - (long)r * C);
        out[(long)e * rc + (long)c * R + r] = (__bf16)in[idx];
    }
}

// ---------------- 3. gating: fp32 logits, top-2, softmax-of-2 ----------------
__global__ void moe_gate(const float* __restrict__ x, const float* __restrict__ wg,
                         int* __restrict__ tok_idx, float* __restrict__ tok_gate,
                         int* __restrict__ counts, float* __restrict__ importance,
                         int* __restrict__ loadcnt) {
    int b = blockIdx.x * blockDim.x + threadIdx.x;
    if (b >= BTOK) return;
    float acc[NEXP];
#pragma unroll
    for (int e = 0; e < NEXP; ++e) acc[e] = 0.0f;
    const float* xr = x + (size_t)b * DIN;
    for (int i = 0; i < DIN; ++i) {
        float xv = xr[i];
#pragma unroll
        for (int e = 0; e < NEXP; ++e) acc[e] += xv * wg[i * NEXP + e];
    }
    int i0 = 0;
    for (int e = 1; e < NEXP; ++e) if (acc[e] > acc[i0]) i0 = e;   // lowest index on ties
    int i1 = (i0 == 0) ? 1 : 0;
    for (int e = 0; e < NEXP; ++e) { if (e == i0) continue; if (acc[e] > acc[i1]) i1 = e; }
    float m = fmaxf(acc[i0], acc[i1]);
    float e0 = __expf(acc[i0] - m), e1 = __expf(acc[i1] - m);
    float inv = 1.0f / (e0 + e1);
    float g0 = e0 * inv, g1 = e1 * inv;
    tok_idx[2*b]   = i0;  tok_idx[2*b+1]  = i1;
    tok_gate[2*b]  = g0;  tok_gate[2*b+1] = g1;
    atomicAdd(&counts[i0], 1);          atomicAdd(&counts[i1], 1);
    atomicAdd(&importance[i0], g0);     atomicAdd(&importance[i1], g1);
    atomicAdd(&loadcnt[i0], 1);         atomicAdd(&loadcnt[i1], 1);
}

// ---------------- 4. padded prefix offsets ----------------
__global__ void moe_offsets(const int* __restrict__ counts, int* __restrict__ offs) {
    if (threadIdx.x == 0 && blockIdx.x == 0) {
        int acc = 0;
        for (int e = 0; e < NEXP; ++e) {
            offs[e] = acc;
            acc += (counts[e] + (BM - 1)) & ~(BM - 1);   // pad to block M-tile
        }
        offs[NEXP] = acc;
    }
}

// ---------------- 5. scatter tokens into expert-grouped rows ----------------
__global__ void moe_scatter(const int* __restrict__ tok_idx, const float* __restrict__ tok_gate,
                            const int* __restrict__ offs, int* __restrict__ cursor,
                            int* __restrict__ token_of_row, float* __restrict__ gate_of_row) {
    int b = blockIdx.x * blockDim.x + threadIdx.x;
    if (b >= BTOK) return;
#pragma unroll
    for (int s = 0; s < TOPK; ++s) {
        int e = tok_idx[2*b + s];
        int pos = atomicAdd(&cursor[e], 1);
        int row = offs[e] + pos;
        token_of_row[row] = b;
        gate_of_row[row]  = tok_gate[2*b + s];
    }
}

// ---------------- 6. GEMM1: H = gelu(Xg @ W1[e] + b1[e])  (WMMA bf16) ----------------
// Block: M=64 x N=256, 8 waves; wave: M=64 x N=32 -> 8 accum.
// Fully-unrolled K with compile-time ping-pong double buffer: prefetch of step s+1
// overlaps the 8 WMMAs of step s, zero register-rotation movs.
__global__ __launch_bounds__(256) void moe_gemm1(
    const __bf16* __restrict__ xbf, const __bf16* __restrict__ W1T,
    const float* __restrict__ bias1, __bf16* __restrict__ H,
    const int* __restrict__ offs, const int* __restrict__ token_of_row) {
    const int rowbase = blockIdx.x * BM;
    if (rowbase >= offs[NEXP]) return;
    int e = 0;
    while (e < NEXP - 1 && rowbase >= offs[e + 1]) ++e;

    const int lane = threadIdx.x & 31;
    const int wave = threadIdx.x >> 5;
    const int lm = lane & 15;
    const int lh = lane >> 4;
    const int n0 = blockIdx.y * 256 + wave * 32;

    const __bf16* arow[4];
#pragma unroll
    for (int mt = 0; mt < 4; ++mt) {
        int token = token_of_row[rowbase + mt * 16 + lm];
        if (token < 0) token = 0;    // padded row: garbage compute, never combined
        arow[mt] = xbf + (size_t)token * DIN + lh * 16;
    }
    const __bf16* brow0 = W1T + ((size_t)e * DH + n0 + lm)      * DIN + lh * 16;
    const __bf16* brow1 = W1T + ((size_t)e * DH + n0 + 16 + lm) * DIN + lh * 16;

    f32x8 acc0[4] = {}, acc1[4] = {};
    bf16x16 a[2][4], b0[2], b1[2];
#pragma unroll
    for (int mt = 0; mt < 4; ++mt) a[0][mt] = load_afrag(arow[mt]);
    b0[0] = *(const bf16x16*)(brow0);
    b1[0] = *(const bf16x16*)(brow1);

    constexpr int NK = DIN / 32;   // 24
#pragma unroll
    for (int s = 0; s < NK; ++s) {
        const int cur = s & 1, nxt = cur ^ 1;
        if (s + 1 < NK) {
            const int k = (s + 1) * 32;
#pragma unroll
            for (int mt = 0; mt < 4; ++mt) a[nxt][mt] = load_afrag(arow[mt] + k);
            b0[nxt] = *(const bf16x16*)(brow0 + k);
            b1[nxt] = *(const bf16x16*)(brow1 + k);
        }
#pragma unroll
        for (int mt = 0; mt < 4; ++mt) {
            acc0[mt] = WMMA_BF16(a[cur][mt], b0[cur], acc0[mt]);
            acc1[mt] = WMMA_BF16(a[cur][mt], b1[cur], acc1[mt]);
        }
    }

    const int na = n0 + lm, nb = n0 + 16 + lm;
    const float biasa = bias1[e * DH + na];
    const float biasb = bias1[e * DH + nb];
#pragma unroll
    for (int mt = 0; mt < 4; ++mt) {
#pragma unroll
        for (int r = 0; r < 8; ++r) {
            int m = rowbase + mt * 16 + r + lh * 8;
            H[(size_t)m * DH + na] = (__bf16)gelu_tanh(acc0[mt][r] + biasa);
            H[(size_t)m * DH + nb] = (__bf16)gelu_tanh(acc1[mt][r] + biasb);
        }
    }
}

// ---------------- 7. GEMM2: O = H @ W2[e] + b2[e]  (WMMA bf16, f32 out) ----------------
__global__ __launch_bounds__(256) void moe_gemm2(
    const __bf16* __restrict__ H, const __bf16* __restrict__ W2T,
    const float* __restrict__ bias2, float* __restrict__ O,
    const int* __restrict__ offs) {
    const int rowbase = blockIdx.x * BM;
    if (rowbase >= offs[NEXP]) return;
    int e = 0;
    while (e < NEXP - 1 && rowbase >= offs[e + 1]) ++e;

    const int lane = threadIdx.x & 31;
    const int wave = threadIdx.x >> 5;
    const int lm = lane & 15;
    const int lh = lane >> 4;
    const int n0 = blockIdx.y * 256 + wave * 32;

    const __bf16* arow[4];
#pragma unroll
    for (int mt = 0; mt < 4; ++mt)
        arow[mt] = H + (size_t)(rowbase + mt * 16 + lm) * DH + lh * 16;
    const __bf16* brow0 = W2T + ((size_t)e * DOUT + n0 + lm)      * DH + lh * 16;
    const __bf16* brow1 = W2T + ((size_t)e * DOUT + n0 + 16 + lm) * DH + lh * 16;

    f32x8 acc0[4] = {}, acc1[4] = {};
    bf16x16 a[2][4], b0[2], b1[2];
#pragma unroll
    for (int mt = 0; mt < 4; ++mt) a[0][mt] = load_afrag(arow[mt]);
    b0[0] = *(const bf16x16*)(brow0);
    b1[0] = *(const bf16x16*)(brow1);

    constexpr int NK = DH / 32;    // 64
#pragma unroll
    for (int s = 0; s < NK; ++s) {
        const int cur = s & 1, nxt = cur ^ 1;
        if (s + 1 < NK) {
            const int k = (s + 1) * 32;
#pragma unroll
            for (int mt = 0; mt < 4; ++mt) a[nxt][mt] = load_afrag(arow[mt] + k);
            b0[nxt] = *(const bf16x16*)(brow0 + k);
            b1[nxt] = *(const bf16x16*)(brow1 + k);
        }
#pragma unroll
        for (int mt = 0; mt < 4; ++mt) {
            acc0[mt] = WMMA_BF16(a[cur][mt], b0[cur], acc0[mt]);
            acc1[mt] = WMMA_BF16(a[cur][mt], b1[cur], acc1[mt]);
        }
    }

    const int na = n0 + lm, nb = n0 + 16 + lm;
    const float biasa = bias2[e * DOUT + na];
    const float biasb = bias2[e * DOUT + nb];
#pragma unroll
    for (int mt = 0; mt < 4; ++mt) {
#pragma unroll
        for (int r = 0; r < 8; ++r) {
            int m = rowbase + mt * 16 + r + lh * 8;
            O[(size_t)m * DOUT + na] = acc0[mt][r] + biasa;
            O[(size_t)m * DOUT + nb] = acc1[mt][r] + biasb;
        }
    }
}

// ---------------- 8. combine: softmax(row) * gate, scatter-add ----------------
__global__ __launch_bounds__(256) void moe_combine(
    const float* __restrict__ O, const int* __restrict__ token_of_row,
    const float* __restrict__ gate_of_row, float* __restrict__ out) {
    int row = blockIdx.x;
    int token = token_of_row[row];
    if (token < 0) return;
    __shared__ float red[256];
    const float* orow = O + (size_t)row * DOUT;
    int t = threadIdx.x;

    float mx = -INFINITY;
    for (int i = t; i < DOUT; i += 256) mx = fmaxf(mx, orow[i]);
    red[t] = mx; __syncthreads();
    for (int s = 128; s > 0; s >>= 1) {
        if (t < s) red[t] = fmaxf(red[t], red[t + s]);
        __syncthreads();
    }
    mx = red[0]; __syncthreads();

    float sum = 0.0f;
    for (int i = t; i < DOUT; i += 256) sum += __expf(orow[i] - mx);
    red[t] = sum; __syncthreads();
    for (int s = 128; s > 0; s >>= 1) {
        if (t < s) red[t] += red[t + s];
        __syncthreads();
    }
    float scale = gate_of_row[row] / red[0];

    float* orow_out = out + (size_t)token * DOUT;
    for (int i = t; i < DOUT; i += 256)
        atomicAdd(&orow_out[i], __expf(orow[i] - mx) * scale);
}

// ---------------- 9. final log with EPS clamp ----------------
__global__ void moe_log(float* __restrict__ out, long n) {
    long stride = (long)gridDim.x * blockDim.x;
    for (long i = blockIdx.x * blockDim.x + threadIdx.x; i < n; i += stride) {
        float v = out[i];
        out[i] = logf(v == 0.0f ? EPS_F : v);
    }
}

// ---------------- 10. aux loss (tiny) ----------------
__global__ void moe_loss(const float* __restrict__ importance, const int* __restrict__ loadcnt,
                         float* __restrict__ out_loss) {
    if (threadIdx.x != 0 || blockIdx.x != 0) return;
    float mi = 0.0f, ml = 0.0f;
    for (int e = 0; e < NEXP; ++e) { mi += importance[e]; ml += (float)loadcnt[e]; }
    mi /= NEXP; ml /= NEXP;
    float vi = 0.0f, vl = 0.0f;
    for (int e = 0; e < NEXP; ++e) {
        float di = importance[e] - mi; vi += di * di;
        float dl = (float)loadcnt[e] - ml; vl += dl * dl;
    }
    vi /= (NEXP - 1); vl /= (NEXP - 1);
    *out_loss = 0.01f * (vi / (mi * mi + 1e-10f) + vl / (ml * ml + 1e-10f));
}

// ---------------- launch ----------------
extern "C" void kernel_launch(void* const* d_in, const int* in_sizes, int n_in,
                              void* d_out, int out_size, void* d_ws, size_t ws_size,
                              hipStream_t stream) {
    const float* x      = (const float*)d_in[0];
    const float* w_gate = (const float*)d_in[1];
    const float* W1     = (const float*)d_in[2];
    const float* b1     = (const float*)d_in[3];
    const float* W2     = (const float*)d_in[4];
    const float* b2     = (const float*)d_in[5];
    float* out = (float*)d_out;

    char* w = (char*)d_ws;
    auto alloc = [&](size_t bytes) -> void* {
        void* p = (void*)w;
        w += (bytes + 255) & ~(size_t)255;
        return p;
    };
    __bf16* xbf  = (__bf16*)alloc((size_t)BTOK * DIN * 2);
    __bf16* W1T  = (__bf16*)alloc((size_t)NEXP * DH * DIN * 2);
    __bf16* W2T  = (__bf16*)alloc((size_t)NEXP * DOUT * DH * 2);
    __bf16* H    = (__bf16*)alloc((size_t)CAP * DH * 2);
    float*  O    = (float*) alloc((size_t)CAP * DOUT * 4);
    int*    tok_idx      = (int*)  alloc((size_t)BTOK * 2 * 4);
    float*  tok_gate     = (float*)alloc((size_t)BTOK * 2 * 4);
    int*    token_of_row = (int*)  alloc((size_t)CAP * 4);
    float*  gate_of_row  = (float*)alloc((size_t)CAP * 4);
    int*    counts       = (int*)  alloc(NEXP * 4);
    int*    cursor       = (int*)  alloc(NEXP * 4);
    float*  importance   = (float*)alloc(NEXP * 4);
    int*    loadcnt      = (int*)  alloc(NEXP * 4);
    int*    offs         = (int*)  alloc((NEXP + 1) * 4);

    const long out_n = (long)BTOK * DOUT;

    moe_init<<<2048, 256, 0, stream>>>(out, (int)(out_n + 1), token_of_row, gate_of_row,
                                       counts, cursor, importance, loadcnt);
    {
        long n = (long)BTOK * DIN;
        cast_bf16<<<(int)((n / 4 + 255) / 256), 256, 0, stream>>>(x, xbf, n);
    }
    {
        long n = (long)NEXP * DIN * DH;
        cast_bf16_t<<<8192, 256, 0, stream>>>(W1, W1T, DIN, DH, n);
    }
    {
        long n = (long)NEXP * DH * DOUT;
        cast_bf16_t<<<8192, 256, 0, stream>>>(W2, W2T, DH, DOUT, n);
    }
    moe_gate<<<BTOK / 256, 256, 0, stream>>>(x, w_gate, tok_idx, tok_gate,
                                             counts, importance, loadcnt);
    moe_offsets<<<1, 1, 0, stream>>>(counts, offs);
    moe_scatter<<<BTOK / 256, 256, 0, stream>>>(tok_idx, tok_gate, offs, cursor,
                                                token_of_row, gate_of_row);
    moe_gemm1<<<dim3(MTILES, DH / 256), 256, 0, stream>>>(xbf, W1T, b1, H, offs, token_of_row);
    moe_gemm2<<<dim3(MTILES, DOUT / 256), 256, 0, stream>>>(H, W2T, b2, O, offs);
    moe_combine<<<CAP, 256, 0, stream>>>(O, token_of_row, gate_of_row, out);
    moe_log<<<4096, 256, 0, stream>>>(out, out_n);
    moe_loss<<<1, 1, 0, stream>>>(importance, loadcnt, out + out_n);
}
